// MaskedAttentionEmb_36721970381326
// MI455X (gfx1250) — matrix operand order repaired and checked by reading, not compiled, not run on hardware
//
#include <hip/hip_runtime.h>
#include <hip/hip_bf16.h>
#include <math.h>

typedef __bf16 bf16;
typedef __attribute__((ext_vector_type(16))) __bf16 v16bf;
typedef __attribute__((ext_vector_type(8)))  __bf16 v8bf;
typedef __attribute__((ext_vector_type(8)))  float  v8f;

constexpr int Bb    = 256;
constexpr int Ss    = 128;
constexpr int HIDd  = 768;
constexpr int LYR   = 12;
constexpr int FFd   = 3072;
constexpr int NHEAD = 12;
constexpr int DHd   = 64;
constexpr int Mrows = Bb * Ss;           // 32768 tokens
constexpr int NVALID = 116;              // rows r <= S - (4*CW+1) = 115

// CDNA5 async global->LDS path (ASYNCcnt-tracked), with safe fallback.
// Builtin signature (from compiler diagnostics): (int4 AS1*, int4 AS3*, imm, imm),
// first parameter is NON-const.
#if defined(__gfx1250__) && __has_builtin(__builtin_amdgcn_global_load_async_to_lds_b128)
#define HAS_ASYNC 1
typedef int v4i_t __attribute__((vector_size(16)));
typedef v4i_t __attribute__((address_space(1)))* gas_p;
typedef v4i_t __attribute__((address_space(3)))* las_p;
#define ASYNC_LD_B128(gptr, lptr)                                        \
  __builtin_amdgcn_global_load_async_to_lds_b128(                        \
      (gas_p)(void*)const_cast<bf16*>((const bf16*)(gptr)),              \
      (las_p)(void*)(lptr), 0, 0)
#if __has_builtin(__builtin_amdgcn_s_wait_asynccnt)
#define ASYNC_WAIT() __builtin_amdgcn_s_wait_asynccnt(0)
#else
#define ASYNC_WAIT() asm volatile("s_wait_asynccnt 0x0" ::: "memory")
#endif
#else
#define HAS_ASYNC 0
#define ASYNC_WAIT()
#endif

#define DEV static __device__ __forceinline__

DEV v16bf frag_contig(const bf16* p) {   // 16 contiguous bf16 (B-matrix fragment)
  v8bf lo = *(const v8bf*)p;
  v8bf hi = *(const v8bf*)(p + 8);
  v16bf r;
#pragma unroll
  for (int i = 0; i < 8; ++i) { r[i] = lo[i]; r[i + 8] = hi[i]; }
  return r;
}
DEV v16bf frag_split(const bf16* p) {    // p[0..7] and p[16..23] (A-matrix fragment)
  v8bf lo = *(const v8bf*)p;
  v8bf hi = *(const v8bf*)(p + 16);
  v16bf r;
#pragma unroll
  for (int i = 0; i < 8; ++i) { r[i] = lo[i]; r[i + 8] = hi[i]; }
  return r;
}
DEV v8f zero_v8f() {
  v8f z;
#pragma unroll
  for (int i = 0; i < 8; ++i) z[i] = 0.f;
  return z;
}
DEV float gelu_exact(float x) { return 0.5f * x * (1.f + erff(x * 0.70710678118654752f)); }

// ---------------------------------------------------------------- fp32 -> bf16
__global__ __launch_bounds__(256) void cvt_bf16_k(const float* __restrict__ src,
                                                  bf16* __restrict__ dst, size_t n) {
  size_t i = (size_t)blockIdx.x * blockDim.x + threadIdx.x;
  size_t stride = (size_t)gridDim.x * blockDim.x;
  for (; i < n; i += stride) dst[i] = (bf16)src[i];
}

// --------------------------------------------------------- generic WMMA GEMM
// C[M,N] = A[M,K](bf16) * W[K,N](bf16) + bias, fused epilogue.
// Double-buffered LDS; async global->LDS staging of A overlapped with WMMA.
enum { EPI_BF16 = 0, EPI_F32 = 1, EPI_GELU_BF16 = 2, EPI_GELU_PE = 3 };

template <int EPI>
__global__ __launch_bounds__(256)
void gemm_bf16_k(const bf16* __restrict__ A, const bf16* __restrict__ W,
                 const float* __restrict__ bias, void* __restrict__ outp,
                 void* __restrict__ out2p, int N, int K) {
  __shared__ bf16 As[2][128 * 40];   // 128 x 32, stride 40 (pad)
  __shared__ bf16 Bt[2][128 * 40];   // transposed W tile: [n][k], stride 40

  const int t = threadIdx.x;
  const int lane = t & 31, wid = t >> 5;
  const int wr = wid >> 2, wc = wid & 3;     // 2 x 4 wave grid
  const int mbase = wr * 64, nbase = wc * 32;
  const int l15 = lane & 15, hi = lane >> 4;
  const size_t rowA0 = (size_t)blockIdx.y * 128;
  const size_t colB0 = (size_t)blockIdx.x * 128;

  const int rA = t >> 1, cA = (t & 1) * 16;
  const int kr = t >> 4, cb = (t & 15) * 8;

  v8f acc[4][2];
#pragma unroll
  for (int i = 0; i < 4; ++i)
#pragma unroll
    for (int j = 0; j < 2; ++j) acc[i][j] = zero_v8f();

  auto stage = [&](int k0, int buf) {
    const bf16* ga = A + (rowA0 + rA) * (size_t)K + k0 + cA;
#if HAS_ASYNC
    ASYNC_LD_B128(ga,     &As[buf][rA * 40 + cA]);
    ASYNC_LD_B128(ga + 8, &As[buf][rA * 40 + cA + 8]);
#else
    const uint4* src = (const uint4*)ga;
    uint4 v0 = src[0], v1 = src[1];
    uint4* dst = (uint4*)&As[buf][rA * 40 + cA];
    dst[0] = v0; dst[1] = v1;
#endif
#pragma unroll
    for (int rep = 0; rep < 32; rep += 16) {
      v8bf v = *(const v8bf*)(W + (size_t)(k0 + kr + rep) * N + colB0 + cb);
#pragma unroll
      for (int j = 0; j < 8; ++j) Bt[buf][(cb + j) * 40 + kr + rep] = v[j];
    }
    if (k0 + 32 < K)  // CDNA5 prefetch of the W tile after next
      __builtin_prefetch(W + (size_t)(k0 + 32 + kr) * N + colB0 + cb, 0, 1);
  };

  stage(0, 0);
  const int nIter = K >> 5;
  for (int it = 0; it < nIter; ++it) {
    const int buf = it & 1;
    ASYNC_WAIT();          // my async stores into LDS are visible
    __syncthreads();       // everyone's staging done; other buffer free
    if (it + 1 < nIter) stage((it + 1) << 5, buf ^ 1);

    v16bf af[4], bfg[2];
#pragma unroll
    for (int i = 0; i < 4; ++i)
      af[i] = frag_split(&As[buf][(mbase + i * 16 + l15) * 40 + hi * 8]);
#pragma unroll
    for (int j = 0; j < 2; ++j)
      bfg[j] = frag_split(&Bt[buf][(nbase + j * 16 + l15) * 40 + hi * 8]);
#pragma unroll
    for (int i = 0; i < 4; ++i)
#pragma unroll
      for (int j = 0; j < 2; ++j)
        acc[i][j] = __builtin_amdgcn_wmma_f32_16x16x32_bf16(
            false, af[i], false, bfg[j], (short)0, acc[i][j], false, false);
  }

  float* outF = (float*)outp;
  bf16* outB = (bf16*)outp;
  bf16* out2B = (bf16*)out2p;
#pragma unroll
  for (int i = 0; i < 4; ++i)
#pragma unroll
    for (int j = 0; j < 2; ++j)
#pragma unroll
      for (int r = 0; r < 8; ++r) {
        size_t gm = rowA0 + mbase + i * 16 + r + hi * 8;
        size_t gn = colB0 + nbase + j * 16 + l15;
        float v = acc[i][j][r] + bias[gn];
        size_t idx = gm * (size_t)N + gn;
        if constexpr (EPI == EPI_BF16) {
          outB[idx] = (bf16)v;
        } else if constexpr (EPI == EPI_F32) {
          outF[idx] = v;
        } else if constexpr (EPI == EPI_GELU_BF16) {
          outB[idx] = (bf16)gelu_exact(v);
        } else {  // EPI_GELU_PE: x = gelu(v) + PE[s, n]
          float gv = gelu_exact(v);
          int s = (int)(gm & (Ss - 1));
          int nev = (int)(gn & ~(size_t)1);
          float ang = (float)s * __expf(-9.2103403719761836f * (float)nev / (float)HIDd);
          float pe = (gn & 1) ? __cosf(ang) : __sinf(ang);
          float o = gv + pe;
          outF[idx] = o;
          out2B[idx] = (bf16)o;
        }
      }
}

// -------------------------------------------------- fused sliding-window attention
// block = (head, batch); 8 waves; dynamic LDS: K, V^T, probs(bf16), scores(f32)
constexpr size_t ATTN_SMEM =
    (size_t)(128 * 72 + 64 * 136 + 128 * 136) * sizeof(bf16) + (size_t)128 * 132 * sizeof(float);

__global__ __launch_bounds__(256)
void attn_k(const bf16* __restrict__ qkv, bf16* __restrict__ ctx) {
  extern __shared__ char smem[];
  bf16* Ks = (bf16*)smem;               // [s][dh]  128 x 64, stride 72
  bf16* Vt = Ks + 128 * 72;             // [dh][s]   64 x 128, stride 136
  bf16* Pb = Vt + 64 * 136;             // probs    128 x 128, stride 136
  float* Sc = (float*)(Pb + 128 * 136); // scores   128 x 128, stride 132

  const int t = threadIdx.x, lane = t & 31, w = t >> 5;
  const int h = blockIdx.x, b = blockIdx.y;
  const int l15 = lane & 15, hi = lane >> 4;
  const size_t base = (size_t)(b * Ss) * (3 * HIDd);

  {  // stage K (straight copy -> async global->LDS)
    int r = t >> 1, half = (t & 1) * 32;
    const bf16* src = qkv + base + (size_t)r * (3 * HIDd) + HIDd + h * DHd + half;
    bf16* dst = &Ks[r * 72 + half];
#if HAS_ASYNC
#pragma unroll
    for (int j = 0; j < 4; ++j) ASYNC_LD_B128(src + 8 * j, dst + 8 * j);
#else
    const uint4* s4 = (const uint4*)src;
    uint4* d4 = (uint4*)dst;
#pragma unroll
    for (int j = 0; j < 4; ++j) d4[j] = s4[j];
#endif
  }
  {  // stage V transposed (manual)
    int s = t >> 1, dbase = (t & 1) * 32;
    const bf16* src = qkv + base + (size_t)s * (3 * HIDd) + 2 * HIDd + h * DHd + dbase;
#pragma unroll
    for (int j0 = 0; j0 < 32; j0 += 8) {
      v8bf v = *(const v8bf*)(src + j0);
#pragma unroll
      for (int j = 0; j < 8; ++j) Vt[(dbase + j0 + j) * 136 + s] = v[j];
    }
  }
  ASYNC_WAIT();
  __syncthreads();

  {  // scores = Q K^T / sqrt(DH); wave w owns query rows [16w, 16w+16)
    v8f acc[8];
#pragma unroll
    for (int nt = 0; nt < 8; ++nt) acc[nt] = zero_v8f();
#pragma unroll
    for (int ks = 0; ks < 2; ++ks) {
      const bf16* qp = qkv + base + (size_t)(16 * w + l15) * (3 * HIDd) + h * DHd + ks * 32 + hi * 8;
      v16bf af = frag_split(qp);
      v16bf bfg[8];                      // batch all B fragments: one wait, 8 WMMAs
#pragma unroll
      for (int nt = 0; nt < 8; ++nt)
        bfg[nt] = frag_split(&Ks[(nt * 16 + l15) * 72 + ks * 32 + hi * 8]);
#pragma unroll
      for (int nt = 0; nt < 8; ++nt)
        acc[nt] = __builtin_amdgcn_wmma_f32_16x16x32_bf16(
            false, af, false, bfg[nt], (short)0, acc[nt], false, false);
    }
#pragma unroll
    for (int nt = 0; nt < 8; ++nt)
#pragma unroll
      for (int r = 0; r < 8; ++r)
        Sc[(16 * w + r + hi * 8) * 132 + nt * 16 + l15] = acc[nt][r] * 0.125f;
  }
  __syncthreads();

  if (t < 128) {  // windowed softmax: cols [r, r+12] minus r+6; rows >115 -> zero
    int row = t;
    if (row < NVALID) {
      float e[13], mx = -3.0e38f, sum = 0.f;
#pragma unroll
      for (int j = 0; j < 13; ++j)
        if (j != 6) mx = fmaxf(mx, Sc[row * 132 + row + j]);
#pragma unroll
      for (int j = 0; j < 13; ++j) {
        e[j] = (j == 6) ? 0.f : __expf(Sc[row * 132 + row + j] - mx);
        sum += e[j];
      }
      float inv = 1.f / sum;
      for (int c = 0; c < 128; ++c) Pb[row * 136 + c] = (bf16)0.f;
#pragma unroll
      for (int j = 0; j < 13; ++j) Pb[row * 136 + row + j] = (bf16)(e[j] * inv);
    } else {
      for (int c = 0; c < 128; ++c) Pb[row * 136 + c] = (bf16)0.f;
    }
  }
  __syncthreads();

  {  // ctx = P V
    v8f acc[4];
#pragma unroll
    for (int nt = 0; nt < 4; ++nt) acc[nt] = zero_v8f();
#pragma unroll
    for (int ks = 0; ks < 4; ++ks) {
      v16bf af = frag_split(&Pb[(16 * w + l15) * 136 + ks * 32 + hi * 8]);
      v16bf bfg[4];
#pragma unroll
      for (int nt = 0; nt < 4; ++nt)
        bfg[nt] = frag_split(&Vt[(nt * 16 + l15) * 136 + ks * 32 + hi * 8]);
#pragma unroll
      for (int nt = 0; nt < 4; ++nt)
        acc[nt] = __builtin_amdgcn_wmma_f32_16x16x32_bf16(
            false, af, false, bfg[nt], (short)0, acc[nt], false, false);
    }
#pragma unroll
    for (int nt = 0; nt < 4; ++nt)
#pragma unroll
      for (int r = 0; r < 8; ++r) {
        int srow = 16 * w + r + hi * 8;
        ctx[(size_t)(b * Ss + srow) * HIDd + h * DHd + nt * 16 + l15] = (bf16)acc[nt][r];
      }
  }
}

// ------------------------------------------------------- residual + LayerNorm(768)
__global__ __launch_bounds__(256)
void resln768_k(float* __restrict__ X, const float* __restrict__ T,
                const float* __restrict__ g, const float* __restrict__ bb,
                bf16* __restrict__ XB) {
  __shared__ float red[256];
  const int t = threadIdx.x;
  const size_t row = blockIdx.x;
  float* x = X + row * HIDd;
  const float* tr = T + row * HIDd;
  float v[3];
#pragma unroll
  for (int k = 0; k < 3; ++k) v[k] = x[t + 256 * k] + tr[t + 256 * k];
  red[t] = v[0] + v[1] + v[2];
  __syncthreads();
  for (int st = 128; st > 0; st >>= 1) { if (t < st) red[t] += red[t + st]; __syncthreads(); }
  float m = red[0] * (1.f / 768.f);
  __syncthreads();
  float d = 0.f;
#pragma unroll
  for (int k = 0; k < 3; ++k) { float q = v[k] - m; d += q * q; }
  red[t] = d;
  __syncthreads();
  for (int st = 128; st > 0; st >>= 1) { if (t < st) red[t] += red[t + st]; __syncthreads(); }
  float inv = rsqrtf(red[0] * (1.f / 768.f) + 1e-5f);
#pragma unroll
  for (int k = 0; k < 3; ++k) {
    int c = t + 256 * k;
    float o = (v[k] - m) * inv * g[c] + bb[c];
    x[c] = o;
    XB[row * HIDd + c] = (bf16)o;
  }
}

// ------------------------------------------------ gather + LayerNorm(128) kernels
template <bool BF16OUT>
__global__ __launch_bounds__(128)
void gather_ln_k(const int* __restrict__ idx, const float* __restrict__ Wt,
                 const float* __restrict__ g, const float* __restrict__ bb,
                 void* __restrict__ outp) {
  __shared__ float red[128];
  const int t = threadIdx.x;
  const size_t row = blockIdx.x;
  const float* src = Wt + (size_t)idx[row] * 128;
  float v = src[t];
  red[t] = v;
  __syncthreads();
  for (int st = 64; st > 0; st >>= 1) { if (t < st) red[t] += red[t + st]; __syncthreads(); }
  float m = red[0] * (1.f / 128.f);
  __syncthreads();
  float q = v - m;
  red[t] = q * q;
  __syncthreads();
  for (int st = 64; st > 0; st >>= 1) { if (t < st) red[t] += red[t + st]; __syncthreads(); }
  float inv = rsqrtf(red[0] * (1.f / 128.f) + 1e-5f);
  float o = q * inv * g[t] + bb[t];
  if (BF16OUT) ((bf16*)outp)[row * 128 + t] = (bf16)o;
  else         ((float*)outp)[row * 128 + t] = o;
}

__global__ __launch_bounds__(128)
void ln128_k(const float* __restrict__ in, const float* __restrict__ g,
             const float* __restrict__ bb, float* __restrict__ out) {
  __shared__ float red[128];
  const int t = threadIdx.x;
  const size_t row = blockIdx.x;
  float v = in[row * 128 + t];
  red[t] = v;
  __syncthreads();
  for (int st = 64; st > 0; st >>= 1) { if (t < st) red[t] += red[t + st]; __syncthreads(); }
  float m = red[0] * (1.f / 128.f);
  __syncthreads();
  float q = v - m;
  red[t] = q * q;
  __syncthreads();
  for (int st = 64; st > 0; st >>= 1) { if (t < st) red[t] += red[t + st]; __syncthreads(); }
  float inv = rsqrtf(red[0] * (1.f / 128.f) + 1e-5f);
  out[row * 128 + t] = q * inv * g[t] + bb[t];
}

// ----------------------------------------------------------------- loss kernels
__global__ __launch_bounds__(32)
void loss_part_k(const float* __restrict__ OZ, const float* __restrict__ PRED,
                 const float* __restrict__ NE, float* __restrict__ part) {
  __shared__ float oz[128];
  __shared__ float dots[24];
  const int t = threadIdx.x;
  const int bi = blockIdx.x;
  const int b = bi / NVALID, i = bi % NVALID;
  const float* ozr = OZ + (size_t)(b * Ss + i) * 128;
#pragma unroll
  for (int k = 0; k < 4; ++k) oz[t + 32 * k] = ozr[t + 32 * k];
  __syncthreads();
  if (t < 24) {
    int m6 = t % 6;
    int c = i + m6 + ((t % 12) >= 6 ? 7 : 0);           // window cols, hole at +6
    const float* src = (t < 12 ? PRED : NE) + (size_t)(b * Ss + c) * 128;
    float s = 0.f;
    for (int e = 0; e < 128; ++e) s += oz[e] * src[e];
    dots[t] = s;
  }
  __syncthreads();
  if (t == 0) {
    float p0 = 0, p1 = 0, n0 = 0, n1 = 0;
#pragma unroll
    for (int k = 0; k < 6; ++k) { p0 += dots[k]; p1 += dots[6 + k]; n0 += dots[12 + k]; n1 += dots[18 + k]; }
    p0 *= (1.f / 6.f); p1 *= (1.f / 6.f); n0 *= (1.f / 6.f); n1 *= (1.f / 6.f);
    auto sp = [](float x) { return fmaxf(x, 0.f) + log1pf(__expf(-fabsf(x))); };
    part[bi] = sp(-p0) + sp(-p1) + sp(n0) + sp(n1);
  }
}

__global__ __launch_bounds__(256)
void reduce_loss_k(const float* __restrict__ part, int n, float* __restrict__ out, float scale) {
  __shared__ float red[256];
  int t = threadIdx.x;
  float s = 0.f;
  for (int i = t; i < n; i += 256) s += part[i];
  red[t] = s;
  __syncthreads();
  for (int st = 128; st > 0; st >>= 1) { if (t < st) red[t] += red[t + st]; __syncthreads(); }
  if (t == 0) out[0] = red[0] * scale;
}

// ================================================================== launcher
extern "C" void kernel_launch(void* const* d_in, const int* in_sizes, int n_in,
                              void* d_out, int out_size, void* d_ws, size_t ws_size,
                              hipStream_t stream) {
  const int*   seq      = (const int*)d_in[0];
  const int*   negi     = (const int*)d_in[1];
  const float* embW     = (const float*)d_in[2];
  const float* embg     = (const float*)d_in[3];
  const float* embb     = (const float*)d_in[4];
  const float* projW    = (const float*)d_in[5];
  const float* projb    = (const float*)d_in[6];
  const float* qkvW     = (const float*)d_in[7];
  const float* qkvb     = (const float*)d_in[8];
  const float* aoW      = (const float*)d_in[9];
  const float* aob      = (const float*)d_in[10];
  const float* ln1g     = (const float*)d_in[11];
  const float* ln1b     = (const float*)d_in[12];
  const float* ff1W     = (const float*)d_in[13];
  const float* ff1b     = (const float*)d_in[14];
  const float* ff2W     = (const float*)d_in[15];
  const float* ff2b     = (const float*)d_in[16];
  const float* ln2g     = (const float*)d_in[17];
  const float* ln2b     = (const float*)d_in[18];
  const float* headW    = (const float*)d_in[19];
  const float* headb    = (const float*)d_in[20];
  const float* headg    = (const float*)d_in[21];
  const float* headbeta = (const float*)d_in[22];
  const float* oembW    = (const float*)d_in[23];
  const float* oembg    = (const float*)d_in[24];
  const float* oembb    = (const float*)d_in[25];

  char* wsb = (char*)d_ws;
  size_t off = 0;
  auto alloc = [&](size_t bytes) -> char* {
    char* p = wsb + off;
    off += (bytes + 255) & ~(size_t)255;
    return p;
  };

  bf16*  WBproj = (bf16*)alloc((size_t)128 * HIDd * 2);
  bf16*  WBqkv  = (bf16*)alloc((size_t)LYR * HIDd * 3 * HIDd * 2);
  bf16*  WBao   = (bf16*)alloc((size_t)LYR * HIDd * HIDd * 2);
  bf16*  WBff1  = (bf16*)alloc((size_t)LYR * HIDd * FFd * 2);
  bf16*  WBff2  = (bf16*)alloc((size_t)LYR * FFd * HIDd * 2);
  bf16*  WBhead = (bf16*)alloc((size_t)HIDd * 128 * 2);
  bf16*  ZB     = (bf16*)alloc((size_t)Mrows * 128 * 2);
  float* X      = (float*)alloc((size_t)Mrows * HIDd * 4);
  bf16*  XB     = (bf16*)alloc((size_t)Mrows * HIDd * 2);
  bf16*  QKV    = (bf16*)alloc((size_t)Mrows * 3 * HIDd * 2);
  bf16*  CTX    = (bf16*)alloc((size_t)Mrows * HIDd * 2);
  float* T0     = (float*)alloc((size_t)Mrows * HIDd * 4);
  bf16*  Hbuf   = (bf16*)alloc((size_t)Mrows * FFd * 2);
  float* P0     = (float*)alloc((size_t)Mrows * 128 * 4);
  float* PRED   = (float*)alloc((size_t)Mrows * 128 * 4);
  float* OZ     = (float*)alloc((size_t)Mrows * 128 * 4);
  float* NE     = (float*)alloc((size_t)Mrows * 128 * 4);
  float* LPART  = (float*)alloc((size_t)Bb * NVALID * 4);

  // weights -> bf16
  cvt_bf16_k<<<1024, 256, 0, stream>>>(projW, WBproj, (size_t)128 * HIDd);
  cvt_bf16_k<<<8192, 256, 0, stream>>>(qkvW, WBqkv, (size_t)LYR * HIDd * 3 * HIDd);
  cvt_bf16_k<<<8192, 256, 0, stream>>>(aoW, WBao, (size_t)LYR * HIDd * HIDd);
  cvt_bf16_k<<<8192, 256, 0, stream>>>(ff1W, WBff1, (size_t)LYR * HIDd * FFd);
  cvt_bf16_k<<<8192, 256, 0, stream>>>(ff2W, WBff2, (size_t)LYR * FFd * HIDd);
  cvt_bf16_k<<<1024, 256, 0, stream>>>(headW, WBhead, (size_t)HIDd * 128);

  // z = LN(embed[seq]) -> bf16
  gather_ln_k<true><<<Mrows, 128, 0, stream>>>(seq, embW, embg, embb, (void*)ZB);
  // x = gelu(z @ projW + b) + PE  (f32 X, bf16 XB)
  gemm_bf16_k<EPI_GELU_PE><<<dim3(HIDd / 128, Mrows / 128), 256, 0, stream>>>(
      ZB, WBproj, projb, (void*)X, (void*)XB, HIDd, 128);

  for (int l = 0; l < LYR; ++l) {
    gemm_bf16_k<EPI_BF16><<<dim3(3 * HIDd / 128, Mrows / 128), 256, 0, stream>>>(
        XB, WBqkv + (size_t)l * HIDd * 3 * HIDd, qkvb + (size_t)l * 3 * HIDd,
        (void*)QKV, nullptr, 3 * HIDd, HIDd);
    attn_k<<<dim3(NHEAD, Bb), 256, ATTN_SMEM, stream>>>(QKV, CTX);
    gemm_bf16_k<EPI_F32><<<dim3(HIDd / 128, Mrows / 128), 256, 0, stream>>>(
        CTX, WBao + (size_t)l * HIDd * HIDd, aob + (size_t)l * HIDd,
        (void*)T0, nullptr, HIDd, HIDd);
    resln768_k<<<Mrows, 256, 0, stream>>>(X, T0, ln1g + (size_t)l * HIDd,
                                          ln1b + (size_t)l * HIDd, XB);
    gemm_bf16_k<EPI_GELU_BF16><<<dim3(FFd / 128, Mrows / 128), 256, 0, stream>>>(
        XB, WBff1 + (size_t)l * HIDd * FFd, ff1b + (size_t)l * FFd,
        (void*)Hbuf, nullptr, FFd, HIDd);
    gemm_bf16_k<EPI_F32><<<dim3(HIDd / 128, Mrows / 128), 256, 0, stream>>>(
        Hbuf, WBff2 + (size_t)l * FFd * HIDd, ff2b + (size_t)l * HIDd,
        (void*)T0, nullptr, HIDd, FFd);
    resln768_k<<<Mrows, 256, 0, stream>>>(X, T0, ln2g + (size_t)l * HIDd,
                                          ln2b + (size_t)l * HIDd, XB);
  }

  // preds = LN(x @ headW + b)
  gemm_bf16_k<EPI_F32><<<dim3(1, Mrows / 128), 256, 0, stream>>>(
      XB, WBhead, headb, (void*)P0, nullptr, 128, HIDd);
  ln128_k<<<Mrows, 128, 0, stream>>>(P0, headg, headbeta, PRED);
  // out_z / neg_e
  gather_ln_k<false><<<Mrows, 128, 0, stream>>>(seq, oembW, oembg, oembb, (void*)OZ);
  gather_ln_k<false><<<Mrows, 128, 0, stream>>>(negi, oembW, oembg, oembb, (void*)NE);

  // loss: per-(b, row) partials, deterministic reduction
  loss_part_k<<<Bb * NVALID, 32, 0, stream>>>(OZ, PRED, NE, LPART);
  reduce_loss_k<<<1, 256, 0, stream>>>(LPART, Bb * NVALID, (float*)d_out,
                                       1.f / (float)(Bb * NVALID * 4));
}
